// PyramidRoIAlign2_33724083208379
// MI455X (gfx1250) — matrix operand order repaired and checked by reading, not compiled, not run on hardware
//
#include <hip/hip_runtime.h>
#include <hip/hip_bf16.h>

// ---------------------------------------------------------------------------
// PyramidRoIAlign + channel-reduce (nhwc,oc->nohw) + bias + relu for gfx1250.
//
// Stage 1: ROIAlign gather (pyramid ~178MB vs 192MB L2 -> L2-resident) -> bf16
//          feature rows [box][bin(64, padded)][k=1024] in workspace.
// Stage 2: per-box [64x1024]x[1024x256] GEMM with v_wmma_f32_16x16x32_bf16,
//          fp32 accumulate, fused bias+relu store to [n][o][49].
//          Each wave owns a 2x4 grid of 16x16 tiles (32 rows x 64 cols):
//          per K-step, 2 A-frags + 4 B-frags (12 b128 loads) feed 8 WMMAs
//          -> 1.5 loads/WMMA register-level reuse.
// ---------------------------------------------------------------------------

typedef __bf16 v16bf __attribute__((ext_vector_type(16)));
typedef __bf16 v8bf  __attribute__((ext_vector_type(8)));
typedef float  v8f   __attribute__((ext_vector_type(8)));

#define NBOX   512
#define CIN    256
#define KTOT   1024   // 4 levels * 256 channels
#define OUTHW  7
#define NBIN   49     // 7*7
#define MPAD   64     // bins padded to 64 rows (4 M-tiles of 16)
#define NOUT   256

static __device__ __forceinline__ __bf16 to_bf16_rne(float f) {
    unsigned u = __builtin_bit_cast(unsigned, f);
    u += 0x7FFFu + ((u >> 16) & 1u);            // round-to-nearest-even
    unsigned short h = (unsigned short)(u >> 16);
    return __builtin_bit_cast(__bf16, h);
}

// ---------------------------------------------------------------------------
// Weight convert: w_reduce[o=256][c=1024] fp32 -> Wt[o=256][k=1024] bf16.
// N-major so a WMMA B-fragment (lane = N column) reads contiguous K runs.
// ---------------------------------------------------------------------------
__global__ void wt_to_bf16(const float* __restrict__ w, __bf16* __restrict__ wt) {
    const int k = blockIdx.x;        // 0..1023
    const int o = threadIdx.x;       // 0..255
    wt[(size_t)o * KTOT + k] = to_bf16_rne(w[(size_t)o * KTOT + k]);
}

// ---------------------------------------------------------------------------
// ROIAlign (legacy, aligned=False), SRxSR=2x2 sample average per bin.
// grid = (bin 0..63, level 0..3, box 0..511), block = 256 channel threads.
// Writes feat[box][bin][level*256 + c] as bf16; bins >= 49 are zero padding.
// ---------------------------------------------------------------------------
__global__ void roialign_to_bf16(const float* __restrict__ p2,
                                 const float* __restrict__ p3,
                                 const float* __restrict__ p4,
                                 const float* __restrict__ p5,
                                 const float* __restrict__ boxes,
                                 const int*   __restrict__ binds,
                                 __bf16*      __restrict__ feat) {
    const int bin = blockIdx.x;      // 0..63
    const int lvl = blockIdx.y;      // 0..3
    const int box = blockIdx.z;      // 0..511
    const int c   = threadIdx.x;     // 0..255

    __bf16* dst = feat + ((size_t)(box * MPAD + bin)) * KTOT + lvl * CIN + c;
    if (bin >= NBIN) {               // zero the pad rows (deterministic ws)
        *dst = to_bf16_rne(0.0f);
        return;
    }

    int H, stride;
    const float* f;
    switch (lvl) {
        case 0:  f = p2; H = 256; stride = 4;  break;
        case 1:  f = p3; H = 128; stride = 8;  break;
        case 2:  f = p4; H = 64;  stride = 16; break;
        default: f = p5; H = 32;  stride = 32; break;
    }
    const float inv = 1.0f / (float)stride;

    const float* bx = boxes + box * 4;
    const float x1 = bx[0] * inv, y1 = bx[1] * inv;
    const float x2 = bx[2] * inv, y2 = bx[3] * inv;
    const float bw = fmaxf(x2 - x1, 1.0f) * (1.0f / 7.0f);
    const float bh = fmaxf(y2 - y1, 1.0f) * (1.0f / 7.0f);

    const int by  = bin / OUTHW;
    const int bxi = bin % OUTHW;
    const int bi  = binds[box];
    const float* base = f + ((size_t)bi * CIN + c) * (size_t)H * (size_t)H;
    const float hm1 = (float)(H - 1);

    float acc = 0.0f;
#pragma unroll
    for (int s = 0; s < 4; ++s) {
        const int sy = s >> 1, sx = s & 1;
        const float offy = (float)by  + ((float)sy + 0.5f) * 0.5f;
        const float offx = (float)bxi + ((float)sx + 0.5f) * 0.5f;
        const float ys = fminf(fmaxf(y1 + offy * bh, 0.0f), hm1);
        const float xs = fminf(fmaxf(x1 + offx * bw, 0.0f), hm1);
        const int iy0 = (int)floorf(ys);
        const int ix0 = (int)floorf(xs);
        const int iy1 = min(iy0 + 1, H - 1);
        const int ix1 = min(ix0 + 1, H - 1);
        const float wy = ys - (float)iy0;
        const float wx = xs - (float)ix0;
        const float* r0 = base + (size_t)iy0 * H;
        const float* r1 = base + (size_t)iy1 * H;
        const float v00 = r0[ix0], v01 = r0[ix1];
        const float v10 = r1[ix0], v11 = r1[ix1];
        acc += (v00 * (1.0f - wx) + v01 * wx) * (1.0f - wy) +
               (v10 * (1.0f - wx) + v11 * wx) * wy;
    }
    *dst = to_bf16_rne(acc * 0.25f);
}

// ---------------------------------------------------------------------------
// Reduce GEMM: out[n,o,bin] = relu(sum_k feat[n,bin,k] * Wt[k,o] + b[o]).
//
// One wave computes a 32x64 output block = 2 (M) x 4 (N) tiles of 16x16.
// K loop: per 32-wide K step, 8 v_wmma_f32_16x16x32_bf16 fed by 12 b128
// loads (A frags shared across 4 N-tiles, B frags shared across 2 M-tiles).
//
// ISA 7.12.2 16-bit layouts (wave32):
//   A (16x32): lanes 0-15 row M=lane, K={0..7,16..23}; lanes 16-31 row
//              M=lane-16, K={8..15,24..31}  -> two aligned b128 loads/lane.
//   B (32x16): lanes 0-15 col N=lane, K={0..7,16..23}; lanes 16-31 col
//              N=lane-16, K={8..15,24..31}      -> two aligned b128 loads/lane.
//   C/D:       VGPR r -> M = 8*half + r, lane -> N.
// ---------------------------------------------------------------------------
__global__ void reduce_gemm_wmma(const __bf16* __restrict__ feat, // [512*64*1024]
                                 const __bf16* __restrict__ wt,   // [1024*256]
                                 const float*  __restrict__ bias, // [256]
                                 float*        __restrict__ out)  // [512*256*49]
{
    const int wave   = threadIdx.x >> 5;
    const int tile   = blockIdx.x * 8 + wave;  // 0..4095
    const int box    = tile >> 3;              // 0..511
    const int rem    = tile & 7;
    const int mgroup = rem >> 2;               // 0..1 -> rows mgroup*32..+31
    const int ngroup = rem & 3;                // 0..3 -> cols ngroup*64..+63

    const int lane = threadIdx.x & 31;
    const int half = lane >> 4;                // 0 or 1
    const int lm   = lane & 15;

    // A: lane feeds row (mgroup*32 + m*16 + lm); K chunks start at half*8.
    const __bf16* arow0 = feat + ((size_t)(box * MPAD + mgroup * 32 + lm)) * KTOT
                               + half * 8;
    const __bf16* arow1 = arow0 + (size_t)16 * KTOT;
    // B: lane feeds N column (ngroup*64 + n*16 + lm); K chunks start at half*8.
    const __bf16* brow  = wt + ((size_t)(ngroup * 64 + lm)) * KTOT + half * 8;

    v8f acc[2][4] = {};
#pragma unroll 2
    for (int k0 = 0; k0 < KTOT; k0 += 32) {
        __builtin_prefetch(arow0 + k0 + 64, 0, 1);   // global_prefetch_b8
        __builtin_prefetch(arow1 + k0 + 64, 0, 1);

        v16bf a[2], b[4];
        {
            const v8bf a0lo = *(const v8bf*)(arow0 + k0);
            const v8bf a0hi = *(const v8bf*)(arow0 + k0 + 16);
            const v8bf a1lo = *(const v8bf*)(arow1 + k0);
            const v8bf a1hi = *(const v8bf*)(arow1 + k0 + 16);
#pragma unroll
            for (int i = 0; i < 8; ++i) {
                a[0][i] = a0lo[i];  a[0][8 + i] = a0hi[i];
                a[1][i] = a1lo[i];  a[1][8 + i] = a1hi[i];
            }
        }
#pragma unroll
        for (int n = 0; n < 4; ++n) {
            const v8bf blo = *(const v8bf*)(brow + (size_t)n * 16 * KTOT + k0);
            const v8bf bhi = *(const v8bf*)(brow + (size_t)n * 16 * KTOT + k0 + 16);
#pragma unroll
            for (int i = 0; i < 8; ++i) { b[n][i] = blo[i]; b[n][8 + i] = bhi[i]; }
        }
#pragma unroll
        for (int m = 0; m < 2; ++m) {
#pragma unroll
            for (int n = 0; n < 4; ++n) {
                // 8 args: (neg_a, A, neg_b, B, c_mod, C, reuse_a, reuse_b)
                acc[m][n] = __builtin_amdgcn_wmma_f32_16x16x32_bf16(
                    false, a[m], false, b[n], (short)0, acc[m][n], false, false);
            }
        }
    }

    // Epilogue: bias + relu, scatter to out[n][o][bin] (bins contiguous/lane).
#pragma unroll
    for (int n = 0; n < 4; ++n) {
        const int   o  = ngroup * 64 + n * 16 + lm;   // N index for this lane
        const float bo = bias[o];
        float* outp = out + ((size_t)(box * NOUT + o)) * NBIN;
#pragma unroll
        for (int m = 0; m < 2; ++m) {
            const int binbase = (mgroup * 2 + m) * 16 + half * 8;
#pragma unroll
            for (int r = 0; r < 8; ++r) {
                const int bin = binbase + r;
                if (bin < NBIN) {
                    const float v = acc[m][n][r] + bo;
                    outp[bin] = v > 0.0f ? v : 0.0f;
                }
            }
        }
    }
}

// ---------------------------------------------------------------------------
// Launch. Workspace layout:
//   [0, 512KB)            : Wt bf16 [256][1024]
//   [512KB, 512KB+64MB)   : feat bf16 [512][64][1024]
// ---------------------------------------------------------------------------
extern "C" void kernel_launch(void* const* d_in, const int* in_sizes, int n_in,
                              void* d_out, int out_size, void* d_ws, size_t ws_size,
                              hipStream_t stream) {
    const float* p2    = (const float*)d_in[0];
    const float* p3    = (const float*)d_in[1];
    const float* p4    = (const float*)d_in[2];
    const float* p5    = (const float*)d_in[3];
    const float* boxes = (const float*)d_in[4];
    const int*   binds = (const int*)  d_in[5];
    const float* w     = (const float*)d_in[6];
    const float* bias  = (const float*)d_in[7];
    float* out = (float*)d_out;

    __bf16* wt   = (__bf16*)d_ws;
    __bf16* feat = (__bf16*)((char*)d_ws + (size_t)KTOT * NOUT * sizeof(__bf16));

    // 1) weight -> N-major bf16
    wt_to_bf16<<<KTOT, NOUT, 0, stream>>>(w, wt);

    // 2) ROIAlign gather -> bf16 feature rows (padded to 64 bins)
    roialign_to_bf16<<<dim3(MPAD, 4, NBOX), CIN, 0, stream>>>(
        p2, p3, p4, p5, boxes, binds, feat);

    // 3) WMMA reduce GEMM + bias + relu
    //    waves = 512 boxes * 2 mgroups * 4 ngroups = 4096; 8 waves/block.
    reduce_gemm_wmma<<<4096 / 8, 256, 0, stream>>>(feat, wt, bias, out);
}
